// SNN_53051436040711
// MI455X (gfx1250) — 8-bit matrix operand order repaired and checked by reading, compiled, not run on hardware
//
#include <hip/hip_runtime.h>
#include <hip/hip_bf16.h>


typedef int v8i __attribute__((ext_vector_type(8)));

#define TW        40
#define E_DIM     4096
#define H1_DIM    16
#define WAVES     4
#define CHUNKS    (E_DIM / 64)
#define W1_STRIDE 4112
#define QSCALE    8128.0f
#define DECAYF    0.4f
#define LIF_TH    0.5f

__device__ __forceinline__ unsigned lvl4pack(float4 v, float a, float b) {
    int l0 = min(40, max(0, (int)fmaf(v.x, a, b)));
    int l1 = min(40, max(0, (int)fmaf(v.y, a, b)));
    int l2 = min(40, max(0, (int)fmaf(v.z, a, b)));
    int l3 = min(40, max(0, (int)fmaf(v.w, a, b)));
    return ((unsigned)l0 | ((unsigned)l1 << 8) | ((unsigned)l2 << 16) |
            ((unsigned)l3 << 24)) | 0x80808080u;
}

__global__ __launch_bounds__(WAVES * 32, 1)
void snn_iu8_wmma(const float* __restrict__ X,
                  const float* __restrict__ SC,
                  const float* __restrict__ W1,
                  const float* __restrict__ B1,
                  const float* __restrict__ Wd1,
                  const float* __restrict__ Bd1,
                  const float* __restrict__ Wd2,
                  const float* __restrict__ Bd2,
                  float* __restrict__ OUT)
{
    __shared__ __align__(16) unsigned char w1q[H1_DIM * W1_STRIDE];
    __shared__ float osT[WAVES][16][17];

    const int tid = threadIdx.x;

    for (int idx = tid; idx < H1_DIM * E_DIM / 4; idx += blockDim.x) {
        const int h  = idx >> 10;
        const int e4 = (idx & 1023) << 2;
        const float4 w = *(const float4*)(W1 + h * E_DIM + e4);
        const int q0 = (int)rintf(w.x * QSCALE) & 255;
        const int q1 = (int)rintf(w.y * QSCALE) & 255;
        const int q2 = (int)rintf(w.z * QSCALE) & 255;
        const int q3 = (int)rintf(w.w * QSCALE) & 255;
        *(int*)(w1q + h * W1_STRIDE + e4) = q0 | (q1 << 8) | (q2 << 16) | (q3 << 24);
    }
    __syncthreads();

    const int lane = tid & 31;
    const int wave = tid >> 5;
    const int ln   = lane & 15;
    const int hi   = lane >> 4;
    const int row  = blockIdx.x * (WAVES * 16) + wave * 16 + ln;

    const float s0 = SC[0], s1 = SC[1];
    const float inv01 = 1.0f / (s1 - s0);
    const float la = 41.0f * inv01;
    const float lb = -41.0f * s0 * inv01;

    const float* xrow = X + (long)row * E_DIM;

    v8i acc[TW];
#pragma unroll
    for (int t = 0; t < TW; ++t)
#pragma unroll
        for (int d = 0; d < 8; ++d) acc[t][d] = 0;

    for (int c = 0; c < CHUNKS; ++c) {
        if (c + 1 < CHUNKS)
            __builtin_prefetch(xrow + (c + 1) * 64 + hi * 8, 0, 1);

        unsigned Lv[8];
        const int ebase = c * 64 + hi * 16;
#pragma unroll
        for (int p = 0; p < 4; ++p) {
            const float4 xa = *(const float4*)(xrow + ebase + (p >> 1) * 32 + (p & 1) * 8);
            const float4 xb = *(const float4*)(xrow + ebase + (p >> 1) * 32 + (p & 1) * 8 + 4);
            Lv[2 * p]     = lvl4pack(xa, la, lb);
            Lv[2 * p + 1] = lvl4pack(xb, la, lb);
        }

        const unsigned char* bp = w1q + ln * W1_STRIDE + c * 64 + hi * 16;
        const int4 bA = *(const int4*)(bp);
        const int4 bB = *(const int4*)(bp + 32);
        v8i Bf;
        Bf[0] = bA.x; Bf[1] = bA.y; Bf[2] = bA.z; Bf[3] = bA.w;
        Bf[4] = bB.x; Bf[5] = bB.y; Bf[6] = bB.z; Bf[7] = bB.w;

#pragma unroll
        for (int t = 0; t < TW; ++t) {
            const unsigned ct = (unsigned)(t + 1) * 0x01010101u;
            v8i Af;
#pragma unroll
            for (int d = 0; d < 8; ++d)
                Af[d] = (int)(((Lv[d] - ct) >> 7) & 0x01010101u);
            acc[t] = __builtin_amdgcn_wmma_i32_16x16x64_iu8(
                 false, Af,  true, Bf, acc[t],
                 false,  false);
        }
    }

    const float b1v  = B1[ln];
    const float invq = 1.0f / QSCALE;
    float mem[8], spk[8], osp[8];
#pragma unroll
    for (int j = 0; j < 8; ++j) { mem[j] = 0.f; spk[j] = 0.f; osp[j] = 0.f; }

#pragma unroll
    for (int t = 0; t < TW; ++t) {
#pragma unroll
        for (int j = 0; j < 8; ++j) {
            const float z = (float)acc[t][j] * invq + b1v;
            mem[j] = mem[j] * (DECAYF * (1.0f - spk[j])) + z;
            spk[j] = (mem[j] - LIF_TH > 0.0f) ? 1.0f : 0.0f;
            osp[j] += spk[j];
        }
    }

#pragma unroll
    for (int j = 0; j < 8; ++j)
        osT[wave][j + 8 * hi][ln] = osp[j] * (1.0f / TW);
    __syncthreads();

    if (lane < 16) {
        float v[16];
#pragma unroll
        for (int j = 0; j < 16; ++j) v[j] = osT[wave][lane][j];

        float hid[8];
#pragma unroll
        for (int k = 0; k < 8; ++k) {
            float s = Bd1[k];
#pragma unroll
            for (int j = 0; j < 16; ++j) s += v[j] * Wd1[k * 16 + j];
            hid[k] = 1.0f / (1.0f + expf(-s));
        }
        const long grow = (long)blockIdx.x * (WAVES * 16) + wave * 16 + lane;
#pragma unroll
        for (int o = 0; o < 16; ++o) {
            float r = Bd2[o];
#pragma unroll
            for (int k = 0; k < 8; ++k) r += hid[k] * Wd2[o * 8 + k];
            OUT[grow * 16 + o] = r;
        }
    }
}

extern "C" void kernel_launch(void* const* d_in, const int* in_sizes, int n_in,
                              void* d_out, int out_size, void* d_ws, size_t ws_size,
                              hipStream_t stream) {
    const float* X   = (const float*)d_in[0];
    const float* SC  = (const float*)d_in[1];
    const float* W1  = (const float*)d_in[2];
    const float* B1  = (const float*)d_in[3];
    const float* Wd1 = (const float*)d_in[4];
    const float* Bd1 = (const float*)d_in[5];
    const float* Wd2 = (const float*)d_in[6];
    const float* Bd2 = (const float*)d_in[7];

    const int Brows  = in_sizes[0] / E_DIM;
    const int blocks = Brows / (WAVES * 16);

    snn_iu8_wmma<<<blocks, WAVES * 32, 0, stream>>>(
        X, SC, W1, B1, Wd1, Bd1, Wd2, Bd2, (float*)d_out);
}
